// DecisionTransformerGPT2Attention_29463475650883
// MI455X (gfx1250) — matrix operand order repaired and checked by reading, not compiled, not run on hardware
//
#include <hip/hip_runtime.h>

// ---------------------------------------------------------------------------
// GPT2 attention forward for MI455X (gfx1250, wave32, WMMA f32_16x16x32_f16)
//   B=2, S=2048, D=1024, H=16, HD=64
// d_out = [attn_output (B*S*D f32)] ++ [attn_weights (B*H*S*S f32)]
// ws: q16, k16, v16 ([B,H,S,HD] f16) + ctx16 ([B,S,D] f16)  = 32 MB
// Wave tile 32x64 (acc[2][4] -> 8 v_wmma per 32-K chunk per wave).
// ---------------------------------------------------------------------------

typedef __attribute__((ext_vector_type(16))) _Float16 v16h;
typedef __attribute__((ext_vector_type(8)))  _Float16 v8h;
typedef __attribute__((ext_vector_type(8)))  float    v8f;
typedef int v4i __attribute__((vector_size(4 * sizeof(int))));

#define BDIM 1024
#define SDIM 2048
#define NH   16
#define HD   64
#define ROWS 4096            // B*S
#define BH   32              // B*H

#define BLK_K 32
#define LSTR  40             // LDS row stride in halves (80B, 16B aligned)

#if defined(__gfx1250__) && __has_builtin(__builtin_amdgcn_global_load_async_to_lds_b128)
#define HAVE_ASYNC_LDS 1
#else
#define HAVE_ASYNC_LDS 0
#endif

#if HAVE_ASYNC_LDS
__device__ __forceinline__ void async_cp16(const void* gsrc, void* ldst) {
  __builtin_amdgcn_global_load_async_to_lds_b128(
      (__attribute__((address_space(1))) v4i*)(void*)gsrc,
      (__attribute__((address_space(3))) v4i*)ldst, 0, 0);
}
__device__ __forceinline__ void async_wait() {
  asm volatile("s_wait_asynccnt 0" ::: "memory");
}
#endif

// ---- fragment loads from LDS (row-major A[M][K], transposed B[N][K]) -------
__device__ __forceinline__ v16h load_a_frag(const _Float16* As, int M, int g) {
  const _Float16* p = As + M * LSTR;
  v8h lo = *(const v8h*)(p + 8 * g);        // K = 8g .. 8g+7
  v8h hi = *(const v8h*)(p + 16 + 8 * g);   // K = 16+8g .. 16+8g+7
  v16h r;
#pragma unroll
  for (int e = 0; e < 8; ++e) { r[e] = lo[e]; r[8 + e] = hi[e]; }
  return r;
}

__device__ __forceinline__ v16h load_b_frag(const _Float16* Bs, int N, int klo) {
  const _Float16* p = Bs + N * LSTR + klo;  // K = klo..klo+7 and 16+klo..16+klo+7
  v8h lo = *(const v8h*)(p);
  v8h hi = *(const v8h*)(p + 16);
  v16h r;
#pragma unroll
  for (int e = 0; e < 8; ++e) { r[e] = lo[e]; r[8 + e] = hi[e]; }
  return r;
}

// ---- one 32x64 wave tile, accumulate over a 32-wide K chunk ----------------
__device__ __forceinline__ void wave_tile_mma(const _Float16* As, const _Float16* Bs,
                                              int rm, int cn, int lane,
                                              v8f acc[2][4]) {
  const int g = lane >> 4, lm = lane & 15;
  v16h a[2], b[4];
  a[0] = load_a_frag(As, rm + lm, g);
  a[1] = load_a_frag(As, rm + 16 + lm, g);
#pragma unroll
  for (int j = 0; j < 4; ++j) b[j] = load_b_frag(Bs, cn + 16 * j + lm, 8 * g);
#pragma unroll
  for (int i = 0; i < 2; ++i)
#pragma unroll
    for (int j = 0; j < 4; ++j)
      acc[i][j] = __builtin_amdgcn_wmma_f32_16x16x32_f16(
          false, a[i], false, b[j], (short)0, acc[i][j], false, false);
}

__device__ __forceinline__ void zero_acc(v8f acc[2][4]) {
  v8f z = {0.f, 0.f, 0.f, 0.f, 0.f, 0.f, 0.f, 0.f};
#pragma unroll
  for (int i = 0; i < 2; ++i)
#pragma unroll
    for (int j = 0; j < 4; ++j) acc[i][j] = z;
}

// ===========================================================================
// Kernel 1: fused QKV projection  C[4096,3072] = X @ Wqkv + b
//   WG tile 128x128, waves 4M x 2N.  -> q16 (pre-scaled 0.125), k16, v16
// ===========================================================================
__global__ __launch_bounds__(256) void qkv_gemm_kernel(
    const float* __restrict__ X, const float* __restrict__ W,
    const float* __restrict__ bias,
    _Float16* __restrict__ q16, _Float16* __restrict__ k16, _Float16* __restrict__ v16) {
  __shared__ __align__(16) _Float16 As[128 * LSTR];
  __shared__ __align__(16) _Float16 Bs[128 * LSTR];

  const int tid = threadIdx.x, lane = tid & 31, w = tid >> 5;
  const int rm = (w >> 1) * 32, cn = (w & 1) * 64;
  const int m0 = blockIdx.y * 128, n0 = blockIdx.x * 128;

  v8f acc[2][4]; zero_acc(acc);

  for (int k0 = 0; k0 < BDIM; k0 += BLK_K) {
    if (k0 + BLK_K < BDIM) {
      __builtin_prefetch(&X[(size_t)m0 * BDIM + k0 + BLK_K], 0, 1);
      __builtin_prefetch(&W[(size_t)(k0 + BLK_K) * (3 * BDIM) + n0], 0, 1);
    }
#pragma unroll
    for (int it = 0; it < 16; ++it) {           // A: 128x32 f32 -> f16
      int idx = it * 256 + tid, r = idx >> 5, c = idx & 31;
      As[r * LSTR + c] = (_Float16)X[(size_t)(m0 + r) * BDIM + k0 + c];
    }
#pragma unroll
    for (int it = 0; it < 16; ++it) {           // B: 32x128 f32 -> f16, transposed
      int idx = it * 256 + tid, kr = idx >> 7, c = idx & 127;
      Bs[c * LSTR + kr] = (_Float16)W[(size_t)(k0 + kr) * (3 * BDIM) + n0 + c];
    }
    __syncthreads();
    wave_tile_mma(As, Bs, rm, cn, lane, acc);
    __syncthreads();
  }

  const int g = lane >> 4, lm = lane & 15;
#pragma unroll
  for (int i = 0; i < 2; ++i)
#pragma unroll
    for (int j = 0; j < 4; ++j)
#pragma unroll
      for (int r = 0; r < 8; ++r) {
        int row = m0 + rm + i * 16 + r + g * 8;
        int col = n0 + cn + j * 16 + lm;
        float v = acc[i][j][r] + bias[col];
        int part = col >> 10, d = col & 1023;
        int h = d >> 6, hd = d & 63;
        int b = row >> 11, s = row & 2047;
        size_t dst = (((size_t)(b * NH + h)) * SDIM + s) * HD + hd;
        if (part == 0)      q16[dst] = (_Float16)(v * 0.125f);
        else if (part == 1) k16[dst] = (_Float16)v;
        else                v16[dst] = (_Float16)v;
      }
}

// ===========================================================================
// Kernel 2: attention scores  S[bh] = q16[bh] @ k16[bh]^T + mask  (f32 out)
//   WG tile 128x128; f16 staging uses async global->LDS when available.
// ===========================================================================
__global__ __launch_bounds__(256) void scores_kernel(
    const _Float16* __restrict__ q16, const _Float16* __restrict__ k16,
    const float* __restrict__ mask, float* __restrict__ attn) {
  const int tid = threadIdx.x;
  const int bh = blockIdx.z, b = bh >> 4;
  const int m0 = blockIdx.y * 128, n0 = blockIdx.x * 128;
  float* out = attn + (size_t)bh * SDIM * SDIM;
  const float* mrow = mask + (size_t)b * SDIM * SDIM;

  if (n0 > m0 + 127) {                          // fully causal-masked tile
#pragma unroll
    for (int it = 0; it < 64; ++it) {
      int idx = it * 256 + tid, r = idx >> 7, c = idx & 127;
      out[(size_t)(m0 + r) * SDIM + n0 + c] = mrow[(size_t)(m0 + r) * SDIM + n0 + c];
    }
    return;
  }

  __shared__ __align__(16) _Float16 As[128 * LSTR];
  __shared__ __align__(16) _Float16 Bs[128 * LSTR];
  const int lane = tid & 31, w = tid >> 5;
  const int rm = (w >> 1) * 32, cn = (w & 1) * 64;
  const _Float16* Q = q16 + (size_t)bh * SDIM * HD;
  const _Float16* K = k16 + (size_t)bh * SDIM * HD;

  v8f acc[2][4]; zero_acc(acc);

  for (int k0 = 0; k0 < HD; k0 += BLK_K) {      // K dim = 64 -> 2 chunks
#if HAVE_ASYNC_LDS
    // A rows: 128 x (32 f16 = 64B) = 4 chunks of 16B per row, 512 total
#pragma unroll
    for (int it = 0; it < 2; ++it) {
      int idx = it * 256 + tid, r = idx >> 2, ch = idx & 3;
      async_cp16(&Q[(size_t)(m0 + r) * HD + k0] + ch * 8,      // 8 halves = 16B
                 &As[r * LSTR] + ch * 8);
    }
    // B rows (transposed layout == direct row copy of K): 128 rows
#pragma unroll
    for (int it = 0; it < 2; ++it) {
      int idx = it * 256 + tid, r = idx >> 2, ch = idx & 3;
      async_cp16(&K[(size_t)(n0 + r) * HD + k0] + ch * 8,
                 &Bs[r * LSTR] + ch * 8);
    }
    async_wait();
#else
#pragma unroll
    for (int it = 0; it < 16; ++it) {
      int idx = it * 256 + tid, r = idx >> 5, c = idx & 31;
      As[r * LSTR + c] = Q[(size_t)(m0 + r) * HD + k0 + c];
    }
#pragma unroll
    for (int it = 0; it < 16; ++it) {
      int idx = it * 256 + tid, kr = idx >> 7, c = idx & 127;
      Bs[c * LSTR + kr] = K[(size_t)(n0 + c) * HD + k0 + kr];
    }
#endif
    __syncthreads();
    wave_tile_mma(As, Bs, rm, cn, lane, acc);
    __syncthreads();
  }

  const int g = lane >> 4, lm = lane & 15;
#pragma unroll
  for (int i = 0; i < 2; ++i)
#pragma unroll
    for (int j = 0; j < 4; ++j)
#pragma unroll
      for (int r = 0; r < 8; ++r) {
        int row = m0 + rm + i * 16 + r + g * 8;
        int col = n0 + cn + j * 16 + lm;
        out[(size_t)row * SDIM + col] = acc[i][j][r] + mrow[(size_t)row * SDIM + col];
      }
}

// ===========================================================================
// Kernel 3: row softmax over attn (in-place), one block per row
// ===========================================================================
__global__ __launch_bounds__(256) void softmax_kernel(float* __restrict__ attn) {
  __shared__ float red[256];
  const int tid = threadIdx.x;
  float* p = attn + (size_t)blockIdx.x * SDIM;

  float v[8];
#pragma unroll
  for (int i = 0; i < 8; ++i) v[i] = p[tid + i * 256];

  float mx = v[0];
#pragma unroll
  for (int i = 1; i < 8; ++i) mx = fmaxf(mx, v[i]);
  red[tid] = mx; __syncthreads();
  for (int s = 128; s > 0; s >>= 1) {
    if (tid < s) red[tid] = fmaxf(red[tid], red[tid + s]);
    __syncthreads();
  }
  mx = red[0]; __syncthreads();

  float sum = 0.f;
#pragma unroll
  for (int i = 0; i < 8; ++i) { v[i] = expf(v[i] - mx); sum += v[i]; }
  red[tid] = sum; __syncthreads();
  for (int s = 128; s > 0; s >>= 1) {
    if (tid < s) red[tid] += red[tid + s];
    __syncthreads();
  }
  const float inv = 1.f / red[0];
#pragma unroll
  for (int i = 0; i < 8; ++i) p[tid + i * 256] = v[i] * inv;
}

// ===========================================================================
// Kernel 4: context = probs @ V  -> ctx16 [B,S,H*HD] f16 (merged heads)
//   WG tile 256x64, waves 8M x 1N.
// ===========================================================================
__global__ __launch_bounds__(256) void pv_kernel(
    const float* __restrict__ attn, const _Float16* __restrict__ v16,
    _Float16* __restrict__ ctx) {
  __shared__ __align__(16) _Float16 As[256 * LSTR];
  __shared__ __align__(16) _Float16 Bs[64 * LSTR];

  const int tid = threadIdx.x, lane = tid & 31, w = tid >> 5;
  const int rm = w * 32, cn = 0;
  const int bh = blockIdx.y, b = bh >> 4, h = bh & 15;
  const int m0 = blockIdx.x * 256;
  const float* P = attn + (size_t)bh * SDIM * SDIM;
  const _Float16* V = v16 + (size_t)bh * SDIM * HD;

  v8f acc[2][4]; zero_acc(acc);

  for (int k0 = 0; k0 < SDIM; k0 += BLK_K) {
    if (k0 > m0 + 255) break;                   // causal: probs are exactly 0
#pragma unroll
    for (int it = 0; it < 32; ++it) {           // A: 256x32 probs f32 -> f16
      int idx = it * 256 + tid, r = idx >> 5, c = idx & 31;
      As[r * LSTR + c] = (_Float16)P[(size_t)(m0 + r) * SDIM + k0 + c];
    }
#pragma unroll
    for (int it = 0; it < 8; ++it) {            // B: V[k][n] transposed stage
      int idx = it * 256 + tid, kr = idx >> 6, c = idx & 63;
      Bs[c * LSTR + kr] = V[(size_t)(k0 + kr) * HD + c];
    }
    __syncthreads();
    wave_tile_mma(As, Bs, rm, cn, lane, acc);
    __syncthreads();
  }

  const int g = lane >> 4, lm = lane & 15;
#pragma unroll
  for (int i = 0; i < 2; ++i)
#pragma unroll
    for (int j = 0; j < 4; ++j)
#pragma unroll
      for (int r = 0; r < 8; ++r) {
        int s = m0 + rm + i * 16 + r + g * 8;
        int col = cn + j * 16 + lm;             // 0..63 within head
        ctx[((size_t)(b * SDIM + s)) * BDIM + h * HD + col] = (_Float16)acc[i][j][r];
      }
}

// ===========================================================================
// Kernel 5: output projection  out[4096,1024] = ctx16 @ Wp + bp
//   WG tile 128x128; A staging (f16) async when available.
// ===========================================================================
__global__ __launch_bounds__(256) void proj_kernel(
    const _Float16* __restrict__ ctx, const float* __restrict__ Wp,
    const float* __restrict__ bp, float* __restrict__ out) {
  __shared__ __align__(16) _Float16 As[128 * LSTR];
  __shared__ __align__(16) _Float16 Bs[128 * LSTR];

  const int tid = threadIdx.x, lane = tid & 31, w = tid >> 5;
  const int rm = (w >> 1) * 32, cn = (w & 1) * 64;
  const int m0 = blockIdx.y * 128, n0 = blockIdx.x * 128;

  v8f acc[2][4]; zero_acc(acc);

  for (int k0 = 0; k0 < BDIM; k0 += BLK_K) {
#if HAVE_ASYNC_LDS
#pragma unroll
    for (int it = 0; it < 2; ++it) {            // A: 128 rows x 64B, async
      int idx = it * 256 + tid, r = idx >> 2, ch = idx & 3;
      async_cp16(&ctx[(size_t)(m0 + r) * BDIM + k0] + ch * 8,
                 &As[r * LSTR] + ch * 8);
    }
#else
#pragma unroll
    for (int it = 0; it < 16; ++it) {
      int idx = it * 256 + tid, r = idx >> 5, c = idx & 31;
      As[r * LSTR + c] = ctx[(size_t)(m0 + r) * BDIM + k0 + c];
    }
#endif
#pragma unroll
    for (int it = 0; it < 16; ++it) {           // B: f32 -> f16, transposed
      int idx = it * 256 + tid, kr = idx >> 7, c = idx & 127;
      Bs[c * LSTR + kr] = (_Float16)Wp[(size_t)(k0 + kr) * BDIM + n0 + c];
    }
#if HAVE_ASYNC_LDS
    async_wait();
#endif
    __syncthreads();
    wave_tile_mma(As, Bs, rm, cn, lane, acc);
    __syncthreads();
  }

  const int g = lane >> 4, lm = lane & 15;
#pragma unroll
  for (int i = 0; i < 2; ++i)
#pragma unroll
    for (int j = 0; j < 4; ++j)
#pragma unroll
      for (int r = 0; r < 8; ++r) {
        int row = m0 + rm + i * 16 + r + g * 8;
        int col = n0 + cn + j * 16 + lm;
        out[(size_t)row * BDIM + col] = acc[i][j][r] + bp[col];
      }
}

// ===========================================================================
extern "C" void kernel_launch(void* const* d_in, const int* in_sizes, int n_in,
                              void* d_out, int out_size, void* d_ws, size_t ws_size,
                              hipStream_t stream) {
  const float* hidden = (const float*)d_in[0];   // [B,S,D]
  const float* amask  = (const float*)d_in[1];   // [B,1,S,S]
  const float* attn_w = (const float*)d_in[2];   // [D,3D]
  const float* attn_b = (const float*)d_in[3];   // [3D]
  const float* proj_w = (const float*)d_in[4];   // [D,D]
  const float* proj_b = (const float*)d_in[5];   // [D]

  float* attn_output  = (float*)d_out;                       // B*S*D
  float* attn_weights = attn_output + (size_t)ROWS * BDIM;   // B*H*S*S

  const size_t HEAD_ELEMS = (size_t)BH * SDIM * HD;          // 4,194,304
  _Float16* q16 = (_Float16*)d_ws;
  _Float16* k16 = q16 + HEAD_ELEMS;
  _Float16* v16 = k16 + HEAD_ELEMS;
  _Float16* ctx = v16 + HEAD_ELEMS;                          // [B*S, D] f16

  dim3 blk(256);

  // 1) QKV projection: 3072/128 x 4096/128
  qkv_gemm_kernel<<<dim3(24, 32), blk, 0, stream>>>(
      hidden, attn_w, attn_b, q16, k16, v16);

  // 2) scores + mask: S/128 x S/128 x BH
  scores_kernel<<<dim3(16, 16, BH), blk, 0, stream>>>(
      q16, k16, amask, attn_weights);

  // 3) softmax: one block per row
  softmax_kernel<<<dim3(BH * SDIM), blk, 0, stream>>>(attn_weights);

  // 4) probs @ V: S/256 x BH   (HD == 64, single n-tile)
  pv_kernel<<<dim3(8, BH), blk, 0, stream>>>(attn_weights, v16, ctx);

  // 5) output projection: 1024/128 x 4096/128
  proj_kernel<<<dim3(8, 32), blk, 0, stream>>>(
      ctx, proj_w, proj_b, attn_output);
}
